// MultiHeadAttentionWithRoPE_47167330844765
// MI455X (gfx1250) — matrix operand order repaired and checked by reading, not compiled, not run on hardware
//
#include <hip/hip_runtime.h>

#define S_LEN 2048
#define D_MODEL 1024
#define N_HEADS 16
#define D_K 64

typedef __attribute__((ext_vector_type(16))) __bf16          v16bf;
typedef __attribute__((ext_vector_type(8)))  float           v8f;
typedef __attribute__((ext_vector_type(16))) unsigned short  v16u;
typedef __attribute__((ext_vector_type(8)))  unsigned short  v8u;

// float -> bf16 bits, round-to-nearest-even
__device__ __forceinline__ unsigned short f2bf(float f) {
    unsigned int x = __builtin_bit_cast(unsigned int, f);
    unsigned int r = x + 0x7FFFu + ((x >> 16) & 1u);
    return (unsigned short)(r >> 16);
}

// B-matrix fragment: 16 contiguous bf16 (32B) per lane
__device__ __forceinline__ v16bf load_b_frag(const unsigned short* p) {
    v16u u = *(const v16u*)p;
    return __builtin_bit_cast(v16bf, u);
}

// A-matrix fragment from two 8-element (16B) chunks
__device__ __forceinline__ v16bf load_a_frag(const unsigned short* p0,
                                             const unsigned short* p1) {
    v8u lo = *(const v8u*)p0;
    v8u hi = *(const v8u*)p1;
    v16u u;
#pragma unroll
    for (int i = 0; i < 8; ++i) { u[i] = lo[i]; u[8 + i] = hi[i]; }
    return __builtin_bit_cast(v16bf, u);
}

// A-matrix fragment from one contiguous 32B chunk
__device__ __forceinline__ v16bf load_a_frag_sw(const unsigned short* p) {
    v16u u = *(const v16u*)p;
    return __builtin_bit_cast(v16bf, u);
}

__device__ __forceinline__ v8f wmma_bf16(v16bf a, v16bf b, v8f c) {
    return __builtin_amdgcn_wmma_f32_16x16x32_bf16(
        false, a, false, b, (short)0, c, false, false);
}

// CDNA5 async copy: 16B/lane global -> LDS, tracked by ASYNCcnt
__device__ __forceinline__ void async_b128(unsigned lds_off, const void* g) {
    asm volatile("global_load_async_to_lds_b128 %0, %1, off"
                 :: "v"(lds_off), "v"(g) : "memory");
}
__device__ __forceinline__ void wait_async0() {
    asm volatile("s_wait_asynccnt 0x0" ::: "memory");
}
// generic shared pointer -> LDS byte offset (LDS aperture: offset = addr[31:0])
__device__ __forceinline__ unsigned lds_off(const void* p) {
    return (unsigned)(unsigned long long)p;
}

// ---------------------------------------------------------------- conversion
__global__ __launch_bounds__(256) void cvt_bf16_kernel(
    const float* __restrict__ in, unsigned short* __restrict__ out, int n) {
    int i = blockIdx.x * blockDim.x + threadIdx.x;
    if (i < n) out[i] = f2bf(in[i]);
}

// ---------------------------------------------------------------- GEMM
// Y[s,e] = sum_d X[s,d] * W[e,d] + bias[e], wave computes 32x64 tile.
// MODE 0: Q -> RoPE, *1/sqrt(dk), bf16 [S][D]   MODE 1: K -> RoPE, bf16 [S][D]
// MODE 2: V -> bf16 transposed [D][S]           MODE 3: O -> +bias, f32 [S][D]
template <int MODE>
__global__ __launch_bounds__(256) void gemm_kernel(
    const unsigned short* __restrict__ X, const unsigned short* __restrict__ W,
    const float* __restrict__ bias, void* __restrict__ outp) {
    const int lane  = threadIdx.x & 31;
    const int lcol  = lane & 15;
    const int lhalf = lane >> 4;
    const int wave  = blockIdx.x * (blockDim.x >> 5) + (threadIdx.x >> 5);
    const int NT    = D_MODEL / 64;          // 16 n-tiles
    const int Mbase = (wave / NT) * 32;
    const int Nbase = (wave % NT) * 64;

    v8f acc[2][4];
#pragma unroll
    for (int mt = 0; mt < 2; ++mt)
#pragma unroll
        for (int j = 0; j < 4; ++j) acc[mt][j] = (v8f)0.0f;

    const unsigned short* arow0 = X + (size_t)(Mbase + lcol) * D_MODEL;
    const unsigned short* arow1 = X + (size_t)(Mbase + 16 + lcol) * D_MODEL;
    const int ah = lhalf ? 16 : 0;
    for (int k = 0; k < D_MODEL; k += 32) {
        __builtin_prefetch(arow0 + k + 256, 0, 1);
        v16bf a0 = load_a_frag(arow0 + k + ah, arow0 + k + ah + 8);
        v16bf a1 = load_a_frag(arow1 + k + ah, arow1 + k + ah + 8);
#pragma unroll
        for (int j = 0; j < 4; ++j) {
            const unsigned short* bp =
                W + (size_t)(Nbase + 16 * j + lcol) * D_MODEL + k + (lhalf ? 16 : 0);
            v16bf b = load_b_frag(bp);
            acc[0][j] = wmma_bf16(a0, b, acc[0][j]);
            acc[1][j] = wmma_bf16(a1, b, acc[1][j]);
        }
    }

    float bcol[4];
#pragma unroll
    for (int j = 0; j < 4; ++j) bcol[j] = bias[Nbase + 16 * j + lcol];

#pragma unroll
    for (int mt = 0; mt < 2; ++mt) {
#pragma unroll
        for (int r = 0; r < 8; ++r) {
            const int srow = Mbase + mt * 16 + r + 8 * lhalf;
            float v[4];
#pragma unroll
            for (int j = 0; j < 4; ++j) v[j] = acc[mt][j][r] + bcol[j];

            if (MODE == 0 || MODE == 1) {
                // RoPE: N-tile == one head's 64 dims; partner d+-32 is acc[j^2]
                float o[4];
#pragma unroll
                for (int j = 0; j < 4; ++j) {
                    const int d = 16 * j + lcol;
                    const int i = d & 31;
                    const float inv_freq =
                        __expf(-(float)i * 0.28782313662425574f);
                    float sn, cs;
                    __sincosf((float)srow * inv_freq, &sn, &cs);
                    const float partner = v[j ^ 2];
                    o[j] = (j < 2) ? (v[j] * cs - partner * sn)
                                   : (v[j] * cs + partner * sn);
                }
                unsigned short* Y = (unsigned short*)outp;
                const float scl = (MODE == 0) ? 0.125f : 1.0f;
#pragma unroll
                for (int j = 0; j < 4; ++j)
                    Y[(size_t)srow * D_MODEL + Nbase + 16 * j + lcol] =
                        f2bf(o[j] * scl);
            } else if (MODE == 2) {
                unsigned short* Vt = (unsigned short*)outp;  // [D][S]
#pragma unroll
                for (int j = 0; j < 4; ++j)
                    Vt[(size_t)(Nbase + 16 * j + lcol) * S_LEN + srow] =
                        f2bf(v[j]);
            } else {
                float* Y = (float*)outp;
#pragma unroll
                for (int j = 0; j < 4; ++j)
                    Y[(size_t)srow * D_MODEL + Nbase + 16 * j + lcol] = v[j];
            }
        }
    }
}

// ---------------------------------------------------------------- attention
// Block = 8 waves, one head, 128 query rows. K/V chunks (64 keys) staged to
// LDS with async-to-LDS double buffering; online softmax per wave.
__global__ __launch_bounds__(256) void attn_kernel(
    const unsigned short* __restrict__ Q, const unsigned short* __restrict__ K,
    const unsigned short* __restrict__ Vt, unsigned short* __restrict__ Aout) {
    __shared__ alignas(16) unsigned short Klds[2][64 * 64];  // [key][dim]
    __shared__ alignas(16) unsigned short Vlds[2][64 * 64];  // [dim][key]
    __shared__ alignas(16) unsigned short Plds[8][16 * 64];  // [row][key], linear

    const int tid   = threadIdx.x;
    const int lane  = tid & 31;
    const int lcol  = lane & 15;
    const int lhalf = lane >> 4;
    const int w     = tid >> 5;
    const int h     = blockIdx.x >> 4;   // 16 blocks per head
    const int qg    = blockIdx.x & 15;
    const int Qrow  = (qg * 8 + w) * 16;
    const int hoff  = h * D_K;

    // P columns stored linearly (no group swap) so A-frag reads are contiguous-k
    int pcol[4];
#pragma unroll
    for (int jt = 0; jt < 4; ++jt) {
        const int col = 16 * jt + lcol;
        const int g   = (col >> 3) & 3;
        const int gs  = g;
        pcol[jt] = (col & 32) | (gs << 3) | (col & 7);
    }

    auto stage = [&](int jb, int b) {
#pragma unroll
        for (int rep = 0; rep < 2; ++rep) {
            const int i   = tid + rep * 256;  // 0..511 -> 512 x 16B per matrix
            const int row = i >> 3;
            const int c8  = (i & 7) * 8;
            const unsigned short* gk =
                K + (size_t)(jb + row) * D_MODEL + hoff + c8;
            async_b128(lds_off(&Klds[b][row * 64 + c8]), gk);
            const unsigned short* gv =
                Vt + (size_t)(hoff + row) * S_LEN + jb + c8;
            async_b128(lds_off(&Vlds[b][row * 64 + c8]), gv);
        }
    };

    float m_run[8], l_run[8];
    v8f O[4];
#pragma unroll
    for (int r = 0; r < 8; ++r) { m_run[r] = -1e30f; l_run[r] = 0.0f; }
#pragma unroll
    for (int j = 0; j < 4; ++j) O[j] = (v8f)0.0f;

    // Q fragments (K = 64 -> two 16x16x32 steps)
    v16bf qa[2];
#pragma unroll
    for (int ks = 0; ks < 2; ++ks) {
        const unsigned short* qp = Q + (size_t)(Qrow + lcol) * D_MODEL + hoff +
                                   ks * 32 + (lhalf ? 16 : 0);
        qa[ks] = load_a_frag(qp, qp + 8);
    }

    stage(0, 0);

    for (int jb = 0; jb < S_LEN; jb += 64) {
        const int buf = (jb >> 6) & 1;
        wait_async0();          // own async copies into `buf` complete
        __syncthreads();        // everyone's copies visible, prev reads retired
        if (jb + 64 < S_LEN) stage(jb + 64, buf ^ 1);

        // ---- scores: 16x64 chunk (scale folded into Q)
        v8f sc[4];
#pragma unroll
        for (int jt = 0; jt < 4; ++jt) sc[jt] = (v8f)0.0f;
#pragma unroll
        for (int jt = 0; jt < 4; ++jt) {
#pragma unroll
            for (int ks = 0; ks < 2; ++ks) {
                const unsigned short* kp =
                    &Klds[buf][(16 * jt + lcol) * 64 + ks * 32 + lhalf * 16];
                sc[jt] = wmma_bf16(qa[ks], load_b_frag(kp), sc[jt]);
            }
        }
        // ---- online softmax update (row = C-layout element r)
#pragma unroll
        for (int r = 0; r < 8; ++r) {
            float mx = fmaxf(fmaxf(sc[0][r], sc[1][r]),
                             fmaxf(sc[2][r], sc[3][r]));
#pragma unroll
            for (int msk = 1; msk <= 8; msk <<= 1)
                mx = fmaxf(mx, __shfl_xor(mx, msk, 32));
            const float mnew = fmaxf(m_run[r], mx);
            const float fac  = __expf(m_run[r] - mnew);
            m_run[r] = mnew;
            float psum = 0.0f;
#pragma unroll
            for (int jt = 0; jt < 4; ++jt) {
                const float p = __expf(sc[jt][r] - mnew);
                psum += p;
                Plds[w][(r + 8 * lhalf) * 64 + pcol[jt]] = f2bf(p);
            }
#pragma unroll
            for (int msk = 1; msk <= 8; msk <<= 1)
                psum += __shfl_xor(psum, msk, 32);
            l_run[r] = l_run[r] * fac + psum;
#pragma unroll
            for (int jd = 0; jd < 4; ++jd) O[jd][r] *= fac;
        }
        // ---- O += P * V (intra-wave LDS relayout, linear A reads)
#pragma unroll
        for (int ks = 0; ks < 2; ++ks) {
            v16bf pa =
                load_a_frag_sw(&Plds[w][lcol * 64 + ks * 32 + lhalf * 16]);
#pragma unroll
            for (int jd = 0; jd < 4; ++jd) {
                const unsigned short* vp =
                    &Vlds[buf][(16 * jd + lcol) * 64 + ks * 32 + lhalf * 16];
                O[jd] = wmma_bf16(pa, load_b_frag(vp), O[jd]);
            }
        }
    }

    // ---- finalize: divide by row sum, store bf16 [S][D_MODEL]
#pragma unroll
    for (int r = 0; r < 8; ++r) {
        const float inv = 1.0f / l_run[r];
        const int srow  = Qrow + r + 8 * lhalf;
#pragma unroll
        for (int jd = 0; jd < 4; ++jd)
            Aout[(size_t)srow * D_MODEL + hoff + 16 * jd + lcol] =
                f2bf(O[jd][r] * inv);
    }
}

// ---------------------------------------------------------------- launcher
extern "C" void kernel_launch(void* const* d_in, const int* in_sizes, int n_in,
                              void* d_out, int out_size, void* d_ws, size_t ws_size,
                              hipStream_t stream) {
    const float* q_in = (const float*)d_in[0];
    const float* k_in = (const float*)d_in[1];
    const float* v_in = (const float*)d_in[2];
    const float* q_w  = (const float*)d_in[3];
    const float* q_b  = (const float*)d_in[4];
    const float* k_w  = (const float*)d_in[5];
    const float* k_b  = (const float*)d_in[6];
    const float* v_w  = (const float*)d_in[7];
    const float* v_b  = (const float*)d_in[8];
    const float* o_w  = (const float*)d_in[9];
    const float* o_b  = (const float*)d_in[10];

    const size_t ACT = (size_t)S_LEN * D_MODEL;   // 2M elems
    const size_t WEI = (size_t)D_MODEL * D_MODEL; // 1M elems
    char* ws = (char*)d_ws;
    unsigned short* qx_bf = (unsigned short*)(ws + 0);
    unsigned short* kx_bf = (unsigned short*)(ws + 4u  * 1024 * 1024);
    unsigned short* vx_bf = (unsigned short*)(ws + 8u  * 1024 * 1024);
    unsigned short* qw_bf = (unsigned short*)(ws + 12u * 1024 * 1024);
    unsigned short* kw_bf = (unsigned short*)(ws + 14u * 1024 * 1024);
    unsigned short* vw_bf = (unsigned short*)(ws + 16u * 1024 * 1024);
    unsigned short* ow_bf = (unsigned short*)(ws + 18u * 1024 * 1024);
    unsigned short* Qbf   = (unsigned short*)(ws + 20u * 1024 * 1024);
    unsigned short* Kbf   = (unsigned short*)(ws + 24u * 1024 * 1024);
    unsigned short* Vt    = (unsigned short*)(ws + 28u * 1024 * 1024);
    unsigned short* attn  = qx_bf;  // reuse after Q projection consumed

    hipLaunchKernelGGL(cvt_bf16_kernel, dim3((ACT + 255) / 256), dim3(256), 0, stream, q_in, qx_bf, (int)ACT);
    hipLaunchKernelGGL(cvt_bf16_kernel, dim3((ACT + 255) / 256), dim3(256), 0, stream, k_in, kx_bf, (int)ACT);
    hipLaunchKernelGGL(cvt_bf16_kernel, dim3((ACT + 255) / 256), dim3(256), 0, stream, v_in, vx_bf, (int)ACT);
    hipLaunchKernelGGL(cvt_bf16_kernel, dim3((WEI + 255) / 256), dim3(256), 0, stream, q_w, qw_bf, (int)WEI);
    hipLaunchKernelGGL(cvt_bf16_kernel, dim3((WEI + 255) / 256), dim3(256), 0, stream, k_w, kw_bf, (int)WEI);
    hipLaunchKernelGGL(cvt_bf16_kernel, dim3((WEI + 255) / 256), dim3(256), 0, stream, v_w, vw_bf, (int)WEI);
    hipLaunchKernelGGL(cvt_bf16_kernel, dim3((WEI + 255) / 256), dim3(256), 0, stream, o_w, ow_bf, (int)WEI);

    // projections: 64 m-tiles x 16 n-tiles = 1024 waves = 128 blocks
    hipLaunchKernelGGL(gemm_kernel<0>, dim3(128), dim3(256), 0, stream, qx_bf, qw_bf, q_b, (void*)Qbf);
    hipLaunchKernelGGL(gemm_kernel<1>, dim3(128), dim3(256), 0, stream, kx_bf, kw_bf, k_b, (void*)Kbf);
    hipLaunchKernelGGL(gemm_kernel<2>, dim3(128), dim3(256), 0, stream, vx_bf, vw_bf, v_b, (void*)Vt);

    // flash attention: 16 heads x 16 blocks
    hipLaunchKernelGGL(attn_kernel, dim3(256), dim3(256), 0, stream, Qbf, Kbf, Vt, attn);

    // output projection -> f32 d_out
    hipLaunchKernelGGL(gemm_kernel<3>, dim3(128), dim3(256), 0, stream, attn, ow_bf, o_b, d_out);
}
